// MultiHeadSelfAttention_81295140979266
// MI455X (gfx1250) — compile-verified, matrix operand order checked by reading, not run on hardware
//
#include <hip/hip_runtime.h>

// ---- CDNA5 (gfx1250) fused MHSA + projection + LayerNorm ----
// wave32, WMMA bf16 16x16x32, f32 accumulate.
// Pre-pass converts x and weights to bf16 once; QKV k-loop is fully
// unrolled straight-line load->wmma code.  QKV uses a 32x64 tile per wave so
// each B fragment feeds two WMMAs (1.5 b128 loads per wmma).

typedef __attribute__((ext_vector_type(16))) __bf16 v16bf;
typedef __attribute__((ext_vector_type(8)))  float  v8f;

#define NB 4
#define TT 2048
#define CC 1024
#define HH 16
#define DD 64
#define ATT_SCALE 0.125f      // 1/sqrt(64)
#define NEG_BIG  (-3.0e38f)

#define WMMA_BF16(A, B, Cacc) \
  __builtin_amdgcn_wmma_f32_16x16x32_bf16(false, (A), false, (B), (short)0, (Cacc), false, false)

// Hardware convert (v_cvt_pk_bf16_f32).
__device__ __forceinline__ unsigned short bf_bits(float f) {
  union { __bf16 h; unsigned short u; } t;
  t.h = (__bf16)f;
  return t.u;
}

// A/B-type bf16 fragment for v_wmma_*_16x16x32_bf16 from a row-major matrix.
// Lane's row pointer passed in; ISA k-halves: lanes 0-15 take K
// {k0..k0+7, k0+16..k0+23}, lanes 16-31 take K {k0+8..k0+15, k0+24..k0+31}.
__device__ __forceinline__ v16bf frag_bf16(const unsigned short* rowptr, int k0) {
  int half = (threadIdx.x >> 4) & 1;
  const unsigned short* p = rowptr + k0 + half * 8;
  union { uint4 u[2]; v16bf v; } t;
  t.u[0] = *(const uint4*)(p);
  t.u[1] = *(const uint4*)(p + 16);
  return t.v;
}

// ---------------------------------------------------------------------------
// Pre-pass: fp32 -> bf16 (RNE via v_cvt), 4 elements per thread.
// ---------------------------------------------------------------------------
__global__ __launch_bounds__(256) void cvt_bf16_kernel(
    const float* __restrict__ in, unsigned short* __restrict__ out, int n4) {
  int i = blockIdx.x * 256 + threadIdx.x;
  if (i < n4) {
    float4 f = ((const float4*)in)[i];
    ushort4 o;
    o.x = bf_bits(f.x); o.y = bf_bits(f.y);
    o.z = bf_bits(f.z); o.w = bf_bits(f.w);
    ((ushort4*)out)[i] = o;
  }
}

// ---------------------------------------------------------------------------
// Kernel A: K/Q/V projections.  y = x @ W^T  (W [out,in] row-major, so the
// WMMA B-fragment reads W rows directly).  One wave per 32(M)x64(N) tile;
// each B fragment is shared by two A fragments (8 wmma per 12 b128 loads).
// The 64-wide N tile aligns exactly with one head.  k-loop fully unrolled.
// Q,K stored [B,H,T,D] bf16, V stored transposed [B,H,D,T] bf16.
// ---------------------------------------------------------------------------
__global__ __launch_bounds__(256) void qkv_kernel(
    const unsigned short* __restrict__ xb,  const unsigned short* __restrict__ Wkb,
    const unsigned short* __restrict__ Wqb, const unsigned short* __restrict__ Wvb,
    unsigned short* __restrict__ kO, unsigned short* __restrict__ qO,
    unsigned short* __restrict__ vTO) {
  const int lane = threadIdx.x & 31;
  const int half = lane >> 4;
  const int ln16 = lane & 15;
  const int wid  = blockIdx.x * 8 + (threadIdx.x >> 5);
  const int w    = wid / (256 * 16);        // 0=K, 1=Q, 2=V
  const int rem  = wid % (256 * 16);
  const int mt   = rem >> 4;
  const int nt   = rem & 15;

  const unsigned short* W = (w == 0) ? Wkb : (w == 1) ? Wqb : Wvb;
  const int mbase = mt * 32;
  const int nbase = nt * 64;

  const unsigned short* xrow0 = xb + (size_t)(mbase + ln16) * CC;
  const unsigned short* xrow1 = xrow0 + (size_t)16 * CC;
  const unsigned short* wrow[4];
#pragma unroll
  for (int j = 0; j < 4; ++j)
    wrow[j] = W + (size_t)(nbase + 16 * j + ln16) * CC;

  v8f acc0[4] = {};
  v8f acc1[4] = {};
#pragma unroll
  for (int kk = 0; kk < CC; kk += 32) {
    v16bf a0 = frag_bf16(xrow0, kk);
    v16bf a1 = frag_bf16(xrow1, kk);
#pragma unroll
    for (int j = 0; j < 4; ++j) {
      v16bf b = frag_bf16(wrow[j], kk);
      acc0[j] = WMMA_BF16(a0, b, acc0[j]);
      acc1[j] = WMMA_BF16(a1, b, acc1[j]);
    }
  }

  const int h  = nt;              // nbase = nt*64 and D = 64
  const int b  = mbase / TT;      // 32-row tile never crosses a batch boundary
  const int t0 = mbase % TT;
#pragma unroll
  for (int rb = 0; rb < 2; ++rb) {
#pragma unroll
    for (int j = 0; j < 4; ++j) {
      const int d = 16 * j + ln16;
#pragma unroll
      for (int v = 0; v < 8; ++v) {
        const int t = t0 + 16 * rb + v + 8 * half;
        const float fv = rb ? acc1[j][v] : acc0[j][v];
        const unsigned short val = bf_bits(fv);
        if (w == 2) {
          vTO[((size_t)(b * HH + h) * DD + d) * TT + t] = val;
        } else {
          unsigned short* dst = (w == 0) ? kO : qO;
          dst[((size_t)(b * HH + h) * TT + t) * DD + d] = val;
        }
      }
    }
  }
}

// ---------------------------------------------------------------------------
// Kernel B: causal flash-attention.  One wave per (b, h, 16-query tile).
// Computes S^T = K_tile · Q^T so softmax stats are per-lane reductions over
// accumulator VGPRs + one shfl_xor(16), and the S^T accumulator layout maps
// directly onto the bf16 B-fragment layout for the P^T·V WMMA (no shuffles).
// V^T fragment loads are issued BEFORE the softmax VALU block so they overlap.
// ---------------------------------------------------------------------------
__global__ __launch_bounds__(256) void attn_kernel(
    const unsigned short* __restrict__ qM, const unsigned short* __restrict__ kM,
    const unsigned short* __restrict__ vTM, unsigned short* __restrict__ attO) {
  const int lane = threadIdx.x & 31;
  const int half = lane >> 4;
  const int ln16 = lane & 15;
  const int wid  = blockIdx.x * 8 + (threadIdx.x >> 5);
  const int bh   = wid >> 7;                 // / (T/16)
  const int qt   = wid & 127;
  const int b = bh / HH, h = bh % HH;
  const int qbase = qt * 16;

  const unsigned short* qbh = qM  + (size_t)bh * TT * DD;
  const unsigned short* kbh = kM  + (size_t)bh * TT * DD;
  const unsigned short* vbh = vTM + (size_t)bh * DD * TT;

  const unsigned short* qrow = qbh + (size_t)(qbase + ln16) * DD;
  v16bf qf0 = frag_bf16(qrow, 0);
  v16bf qf1 = frag_bf16(qrow, 32);

  v8f o[4] = {};
  float m = NEG_BIG, l = 0.f;
  const int qn  = qbase + ln16;
  const int nkt = (qbase + 15) / 32 + 1;     // causal: skip fully-masked tiles

  for (int it = 0; it < nkt; ++it) {
    const int kb = it * 32;
    const unsigned short* krow0 = kbh + (size_t)(kb + ln16) * DD;
    const unsigned short* krow1 = krow0 + 16 * DD;
    if (it + 1 < nkt) __builtin_prefetch(krow0 + 32 * DD, 0, 0);

    // S^T tiles: [keys 0-15 | 16-31] x [16 queries], K-dim = D = 64 (2 steps)
    v8f s0 = {}, s1 = {};
    s0 = WMMA_BF16(frag_bf16(krow0, 0),  qf0, s0);
    s0 = WMMA_BF16(frag_bf16(krow0, 32), qf1, s0);
    s1 = WMMA_BF16(frag_bf16(krow1, 0),  qf0, s1);
    s1 = WMMA_BF16(frag_bf16(krow1, 32), qf1, s1);

    // Issue V^T loads now; they complete during the softmax math below.
    v16bf vf[4];
#pragma unroll
    for (int j = 0; j < 4; ++j)
      vf[j] = frag_bf16(vbh + (size_t)(16 * j + ln16) * TT + kb, 0);

    float e[16];
    float mloc = NEG_BIG;
#pragma unroll
    for (int v = 0; v < 8; ++v) {
      const int key0 = kb + v + 8 * half;
      const float t0 = (key0      <= qn) ? s0[v] * ATT_SCALE : NEG_BIG;
      const float t1 = (key0 + 16 <= qn) ? s1[v] * ATT_SCALE : NEG_BIG;
      e[v] = t0; e[8 + v] = t1;
      mloc = fmaxf(mloc, fmaxf(t0, t1));
    }
    mloc = fmaxf(mloc, __shfl_xor(mloc, 16, 32));
    const float mnew  = fmaxf(m, mloc);
    const float alpha = __expf(m - mnew);

    float lloc = 0.f;
    union { __bf16 h[16]; v16bf v; } pf;       // P^T B-fragment
#pragma unroll
    for (int i = 0; i < 16; ++i) {
      const float p = __expf(e[i] - mnew);
      lloc += p;
      pf.h[i] = (__bf16)p;
    }
    lloc += __shfl_xor(lloc, 16, 32);
    l = l * alpha + lloc;
    m = mnew;

#pragma unroll
    for (int j = 0; j < 4; ++j)
#pragma unroll
      for (int v = 0; v < 8; ++v) o[j][v] *= alpha;

    // O^T += V^T_tile · P^T   (A = V^T rows contiguous over keys)
#pragma unroll
    for (int j = 0; j < 4; ++j)
      o[j] = WMMA_BF16(vf[j], pf.v, o[j]);
  }

  const float inv = 1.f / l;
  unsigned short* orow = attO + ((size_t)(b * TT + qn) * CC + h * DD);
#pragma unroll
  for (int j = 0; j < 4; ++j)
#pragma unroll
    for (int v = 0; v < 8; ++v)
      orow[16 * j + v + 8 * half] = bf_bits(o[j][v] * inv);
}

// ---------------------------------------------------------------------------
// Kernel C: y = att @ Wp^T, then LayerNorm over the full 1024-wide row.
// 8-wave block owns a complete 16x1024 tile so LN needs no global reduction;
// y staged through 64 KB LDS, row stats via shfl_xor within 16-lane groups.
// k-loop unrolled x4 for scheduling slack (full unroll would spill: 64-VGPR
// accumulator + 8 B-fragments per step).
// ---------------------------------------------------------------------------
__global__ __launch_bounds__(256) void proj_ln_kernel(
    const unsigned short* __restrict__ att, const unsigned short* __restrict__ Wpb,
    const float* __restrict__ gamma, const float* __restrict__ beta,
    float* __restrict__ out) {
  __shared__ float ytile[16 * CC];           // 64 KB
  const int lane = threadIdx.x & 31;
  const int half = lane >> 4;
  const int ln16 = lane & 15;
  const int wv   = threadIdx.x >> 5;
  const int mbase = blockIdx.x * 16;

  const unsigned short* arow = att + (size_t)(mbase + ln16) * CC;
  const unsigned short* wrow[8];
#pragma unroll
  for (int j = 0; j < 8; ++j)
    wrow[j] = Wpb + (size_t)(wv * 128 + 16 * j + ln16) * CC;

  v8f acc[8] = {};
#pragma unroll 4
  for (int kk = 0; kk < CC; kk += 32) {
    v16bf a = frag_bf16(arow, kk);
#pragma unroll
    for (int j = 0; j < 8; ++j) {
      v16bf bf = frag_bf16(wrow[j], kk);
      acc[j] = WMMA_BF16(a, bf, acc[j]);
    }
  }

#pragma unroll
  for (int j = 0; j < 8; ++j) {
    const int col = wv * 128 + 16 * j + ln16;
#pragma unroll
    for (int v = 0; v < 8; ++v)
      ytile[(v + 8 * half) * CC + col] = acc[j][v];
  }
  __syncthreads();

  const int r   = threadIdx.x >> 4;          // row 0..15, 16 threads per row
  const int sub = threadIdx.x & 15;
  float s = 0.f, s2 = 0.f;
  for (int col = sub; col < CC; col += 16) {
    const float yv = ytile[r * CC + col];
    s += yv; s2 += yv * yv;
  }
#pragma unroll
  for (int off = 1; off < 16; off <<= 1) {
    s  += __shfl_xor(s,  off, 32);
    s2 += __shfl_xor(s2, off, 32);
  }
  const float mu   = s * (1.f / CC);
  const float var  = s2 * (1.f / CC) - mu * mu;
  const float rstd = rsqrtf(var + 1e-5f);

  float* orow = out + (size_t)(mbase + r) * CC;
  for (int col = sub; col < CC; col += 16) {
    const float yv = ytile[r * CC + col];
    orow[col] = (yv - mu) * rstd * gamma[col] + beta[col];
  }
}

extern "C" void kernel_launch(void* const* d_in, const int* in_sizes, int n_in,
                              void* d_out, int out_size, void* d_ws, size_t ws_size,
                              hipStream_t stream) {
  const float* x     = (const float*)d_in[0];
  const float* Wk    = (const float*)d_in[1];
  const float* Wq    = (const float*)d_in[2];
  const float* Wv    = (const float*)d_in[3];
  const float* Wp    = (const float*)d_in[4];
  const float* gamma = (const float*)d_in[5];
  const float* beta  = (const float*)d_in[6];
  float* out = (float*)d_out;

  // Workspace layout (bf16 elements):
  //   xb 8M | Wkb/Wqb/Wvb/Wpb 1M each | q,k 8M | vT 8M | att 8M  -> 88 MB
  const size_t nx  = (size_t)NB * TT * CC;        // 8M
  const size_t nw  = (size_t)CC * CC;             // 1M
  const size_t seg = (size_t)NB * HH * TT * DD;   // 8M
  unsigned short* xb  = (unsigned short*)d_ws;
  unsigned short* Wkb = xb  + nx;
  unsigned short* Wqb = Wkb + nw;
  unsigned short* Wvb = Wqb + nw;
  unsigned short* Wpb = Wvb + nw;
  unsigned short* q   = Wpb + nw;
  unsigned short* k   = q   + seg;
  unsigned short* vT  = k   + seg;
  unsigned short* att = vT  + seg;

  // Pre-pass: one-time fp32 -> bf16 conversion (hoists all convert VALU out
  // of the GEMM inner loops; adds ~72 MB of traffic ~ 3 us at 23.3 TB/s).
  cvt_bf16_kernel<<<(int)(nx / 4 / 256), 256, 0, stream>>>(x,  xb,  (int)(nx / 4));
  cvt_bf16_kernel<<<(int)(nw / 4 / 256), 256, 0, stream>>>(Wk, Wkb, (int)(nw / 4));
  cvt_bf16_kernel<<<(int)(nw / 4 / 256), 256, 0, stream>>>(Wq, Wqb, (int)(nw / 4));
  cvt_bf16_kernel<<<(int)(nw / 4 / 256), 256, 0, stream>>>(Wv, Wvb, (int)(nw / 4));
  cvt_bf16_kernel<<<(int)(nw / 4 / 256), 256, 0, stream>>>(Wp, Wpb, (int)(nw / 4));

  // A: 3 weights x 256 M-tiles(32 rows) x 16 N-tiles = 12288 waves -> 1536 blocks
  qkv_kernel<<<1536, 256, 0, stream>>>(xb, Wkb, Wqb, Wvb, k, q, vT);
  // B: B*H*(T/16) = 8192 waves -> 1024 blocks
  attn_kernel<<<1024, 256, 0, stream>>>(q, k, vT, att);
  // C: 8192/16 = 512 row-tiles
  proj_ln_kernel<<<512, 256, 0, stream>>>(att, Wpb, gamma, beta, out);
}
